// ParallelExperts1_58b_63118839382691
// MI455X (gfx1250) — matrix operand order repaired and checked by reading, not compiled, not run on hardware
//
#include <hip/hip_runtime.h>

typedef __attribute__((ext_vector_type(16))) _Float16 v16h;
typedef __attribute__((ext_vector_type(8)))  float    v8f;
typedef __attribute__((ext_vector_type(4)))  _Float16 v4h;

#define NUM_EXPERTS 8
#define IN_SZ   2048
#define OUT_SZ  8192
#define TOK     2048

#define NW ((long long)NUM_EXPERTS * OUT_SZ * IN_SZ)   // 134217728 weight elems
#define NX ((long long)NUM_EXPERTS * TOK * IN_SZ)      // 33554432 input elems

// ---------------- Pass 1: partial sums of |w| (deterministic two-pass) ------
__global__ void pe158_absmean_partial(const float* __restrict__ w,
                                      float* __restrict__ partials,
                                      long long n) {
    __shared__ float sdata[256];
    long long stride = (long long)gridDim.x * blockDim.x;
    float s = 0.0f;
    for (long long i = (long long)blockIdx.x * blockDim.x + threadIdx.x; i < n; i += stride)
        s += fabsf(w[i]);
    sdata[threadIdx.x] = s;
    __syncthreads();
    for (int o = 128; o > 0; o >>= 1) {
        if ((int)threadIdx.x < o) sdata[threadIdx.x] += sdata[threadIdx.x + o];
        __syncthreads();
    }
    if (threadIdx.x == 0) partials[blockIdx.x] = sdata[0];
}

// ---------------- Pass 2: finalize scale -----------------------------------
__global__ void pe158_finalize_scale(const float* __restrict__ partials, int np,
                                     float* __restrict__ scales, float inv_n) {
    __shared__ float sdata[256];
    float s = 0.0f;
    for (int i = threadIdx.x; i < np; i += 256) s += partials[i];
    sdata[threadIdx.x] = s;
    __syncthreads();
    for (int o = 128; o > 0; o >>= 1) {
        if ((int)threadIdx.x < o) sdata[threadIdx.x] += sdata[threadIdx.x + o];
        __syncthreads();
    }
    if (threadIdx.x == 0) {
        float mean = sdata[0] * inv_n;
        float m = fmaxf(mean, 1e-5f);
        scales[0] = 1.0f / m;   // quantization scale
        scales[1] = m;          // dequant multiplier (1/scale)
    }
}

// ---------------- Quantize weights to ternary f16 --------------------------
__global__ void pe158_quant_w(const float* __restrict__ w, _Float16* __restrict__ wq,
                              const float* __restrict__ scales, long long n) {
    const float sc = scales[0];
    long long i = ((long long)blockIdx.x * blockDim.x + threadIdx.x) * 4;
    if (i + 3 < n) {
        float4 v = *(const float4*)(w + i);
        v4h r;
        r.x = (_Float16)fminf(fmaxf(rintf(v.x * sc), -1.0f), 1.0f);
        r.y = (_Float16)fminf(fmaxf(rintf(v.y * sc), -1.0f), 1.0f);
        r.z = (_Float16)fminf(fmaxf(rintf(v.z * sc), -1.0f), 1.0f);
        r.w = (_Float16)fminf(fmaxf(rintf(v.w * sc), -1.0f), 1.0f);
        *(v4h*)(wq + i) = r;
    }
}

// ---------------- Convert activations to f16 -------------------------------
__global__ void pe158_cvt_x(const float* __restrict__ x, _Float16* __restrict__ xh,
                            long long n) {
    long long i = ((long long)blockIdx.x * blockDim.x + threadIdx.x) * 4;
    if (i + 3 < n) {
        float4 v = *(const float4*)(x + i);
        v4h r;
        r.x = (_Float16)v.x; r.y = (_Float16)v.y;
        r.z = (_Float16)v.z; r.w = (_Float16)v.w;
        *(v4h*)(xh + i) = r;
    }
}

// ---------------- Grouped GEMM with WMMA f16 -------------------------------
// Block tile 128(M) x 256(N) x 64(K); 256 threads = 8 wave32 waves.
// Wave grid 2(M) x 4(N): each wave computes 64x64 = 4x4 WMMA 16x16 tiles.
// LDS is double-buffered: one barrier per K-slab.
#define BM 128
#define BN 256
#define BKK 64
#define LDT 72                     // 64 + 8 halves (16B) pad per row
#define ABUF (BM * LDT)            // 9216 halves
#define BBUF (BN * LDT)            // 18432 halves
#define BUFSZ (ABUF + BBUF)        // 27648 halves per buffer

union Frag16 { v16h h; uint4 q[2]; };

__global__ void __launch_bounds__(256)
pe158_moe_wmma_gemm(const _Float16* __restrict__ xh,   // [E, T, I]
                    const _Float16* __restrict__ wq,   // [E, O, I]
                    const float*    __restrict__ scales,
                    float*          __restrict__ out)  // [E, T, O]
{
    __shared__ _Float16 lds[2 * BUFSZ];                // 110,592 B of 320 KB

    const int tid  = threadIdx.x;
    const int lane = tid & 31;
    const int wave = tid >> 5;
    const int wm   = (wave & 1) * 64;   // wave M offset within block tile
    const int wn   = (wave >> 1) * 64;  // wave N offset within block tile

    const int e  = blockIdx.z;
    const int m0 = blockIdx.y * BM;
    const int n0 = blockIdx.x * BN;

    const float inv = scales[1];

    // Cooperative staging: rows of 64 halves = 8 uint4 each.
    // A: 128 rows -> 1024 uint4 -> 4/thread.  B: 256 rows -> 2048 -> 8/thread.
    const int ldrow = tid >> 3;         // 0..31, step 32
    const int ldcol = tid & 7;          // uint4 column in row

    const _Float16* gA = xh + ((size_t)e * TOK    + m0) * IN_SZ + ldcol * 8;
    const _Float16* gB = wq + ((size_t)e * OUT_SZ + n0) * IN_SZ + ldcol * 8;

    uint4 ra[4], rb[8];

    // ---- prologue: stage K-slab 0 into buffer 0 ----
    #pragma unroll
    for (int i = 0; i < 4; ++i)
        ra[i] = *(const uint4*)(gA + (size_t)(ldrow + i * 32) * IN_SZ);
    #pragma unroll
    for (int i = 0; i < 8; ++i)
        rb[i] = *(const uint4*)(gB + (size_t)(ldrow + i * 32) * IN_SZ);
    {
        _Float16* A0 = lds;
        _Float16* B0 = lds + ABUF;
        #pragma unroll
        for (int i = 0; i < 4; ++i)
            *(uint4*)&A0[(ldrow + i * 32) * LDT + ldcol * 8] = ra[i];
        #pragma unroll
        for (int i = 0; i < 8; ++i)
            *(uint4*)&B0[(ldrow + i * 32) * LDT + ldcol * 8] = rb[i];
    }
    __syncthreads();

    v8f acc[4][4];
    #pragma unroll
    for (int mi = 0; mi < 4; ++mi)
        #pragma unroll
        for (int ni = 0; ni < 4; ++ni)
            acc[mi][ni] = (v8f){0.f,0.f,0.f,0.f,0.f,0.f,0.f,0.f};

    const int base0 = (lane >> 4) << 3;   // A: K sub-offset 0 or 8
    const int kb    = (lane >> 4) << 4;   // unused: B also reads with base0 (0 or 8)
    const int mrow  = lane & 15;

    for (int k0 = 0; k0 < IN_SZ; k0 += BKK) {
        const int cur = (k0 >> 6) & 1;
        const _Float16* Ac = lds + cur * BUFSZ;
        const _Float16* Bc = Ac + ABUF;
        const bool more = (k0 + BKK) < IN_SZ;

        // Issue global loads for next K-slab; latency hidden under 256 WMMAs.
        if (more) {
            #pragma unroll
            for (int i = 0; i < 4; ++i)
                ra[i] = *(const uint4*)(gA + (size_t)(ldrow + i * 32) * IN_SZ + (k0 + BKK));
            #pragma unroll
            for (int i = 0; i < 8; ++i)
                rb[i] = *(const uint4*)(gB + (size_t)(ldrow + i * 32) * IN_SZ + (k0 + BKK));
            if (k0 + 2 * BKK < IN_SZ) {   // global_prefetch_b8 for slab after next
                __builtin_prefetch(gA + (size_t)ldrow * IN_SZ + (k0 + 2 * BKK), 0, 1);
                __builtin_prefetch(gB + (size_t)ldrow * IN_SZ + (k0 + 2 * BKK), 0, 1);
            }
        }

        // ---- two 16x16x32 K-steps; each A fragment feeds 4 WMMAs ----
        #pragma unroll
        for (int kk = 0; kk < BKK; kk += 32) {
            Frag16 b[4];
            #pragma unroll
            for (int ni = 0; ni < 4; ++ni) {
                const _Float16* p = &Bc[(wn + ni * 16 + mrow) * LDT + kk + base0];
                b[ni].q[0] = *(const uint4*)(p);       // K = base0 .. base0+7
                b[ni].q[1] = *(const uint4*)(p + 16);  // K = 16+base0 .. 16+base0+7
            }
            #pragma unroll
            for (int mi = 0; mi < 4; ++mi) {
                Frag16 a;
                const _Float16* p = &Ac[(wm + mi * 16 + mrow) * LDT + kk];
                a.q[0] = *(const uint4*)(p + base0);        // K = base0 .. +7
                a.q[1] = *(const uint4*)(p + 16 + base0);   // K = 16+base0 ..
                #pragma unroll
                for (int ni = 0; ni < 4; ++ni)
                    acc[mi][ni] = __builtin_amdgcn_wmma_f32_16x16x32_f16(
                        false, a.h, false, b[ni].h,
                        (short)0, acc[mi][ni], false, false);
            }
        }

        // ---- dump staged registers into the other buffer (no read hazard) ----
        if (more) {
            _Float16* An = lds + (cur ^ 1) * BUFSZ;
            _Float16* Bn = An + ABUF;
            #pragma unroll
            for (int i = 0; i < 4; ++i)
                *(uint4*)&An[(ldrow + i * 32) * LDT + ldcol * 8] = ra[i];
            #pragma unroll
            for (int i = 0; i < 8; ++i)
                *(uint4*)&Bn[(ldrow + i * 32) * LDT + ldcol * 8] = rb[i];
        }
        __syncthreads();   // single barrier per K-slab
    }

    // ---- store D tiles, fold in dequant scale ----
    const int crow = (lane >> 4) << 3;   // 0 or 8
    const int ccol = lane & 15;
    #pragma unroll
    for (int mi = 0; mi < 4; ++mi) {
        #pragma unroll
        for (int ni = 0; ni < 4; ++ni) {
            #pragma unroll
            for (int r = 0; r < 8; ++r) {
                int row = m0 + wm + mi * 16 + crow + r;
                int col = n0 + wn + ni * 16 + ccol;
                out[((size_t)e * TOK + row) * (size_t)OUT_SZ + col] = acc[mi][ni][r] * inv;
            }
        }
    }
}

// ---------------------------------------------------------------------------
extern "C" void kernel_launch(void* const* d_in, const int* in_sizes, int n_in,
                              void* d_out, int out_size, void* d_ws, size_t ws_size,
                              hipStream_t stream) {
    const float* x = (const float*)d_in[0];   // [E*T, I] fp32
    const float* w = (const float*)d_in[1];   // [E, O, I] fp32
    float* out = (float*)d_out;               // [E*T, O] fp32

    const size_t WQ_BYTES = (size_t)NW * sizeof(_Float16);   // 268435456
    const size_t XH_BYTES = (size_t)NX * sizeof(_Float16);   //  67108864

    char* ws = (char*)d_ws;
    _Float16* wq       = (_Float16*)ws;
    _Float16* xh       = (_Float16*)(ws + WQ_BYTES);
    float*    partials = (float*)   (ws + WQ_BYTES + XH_BYTES);
    float*    scales   = partials + 1024;

    pe158_absmean_partial<<<1024, 256, 0, stream>>>(w, partials, NW);
    pe158_finalize_scale<<<1, 256, 0, stream>>>(partials, 1024, scales,
                                                1.0f / (float)NW);
    pe158_quant_w<<<(int)(NW / 1024), 256, 0, stream>>>(w, wq, scales, NW);
    pe158_cvt_x<<<(int)(NX / 1024), 256, 0, stream>>>(x, xh, NX);

    dim3 grid(OUT_SZ / BN, TOK / BM, NUM_EXPERTS);   // (32, 16, 8)
    pe158_moe_wmma_gemm<<<grid, 256, 0, stream>>>(xh, wq, scales, out);
}
